// TEM_31198642438207
// MI455X (gfx1250) — matrix operand order repaired and checked by reading, not compiled, not run on hardware
//
#include <hip/hip_runtime.h>
#include <hip/hip_bf16.h>

// ---------------------------------------------------------------------------
// TEM module for MI455X (gfx1250, wave32, WMMA f32_16x16x32_f16).
// Heavy per-pixel channel mixes run on v_wmma_f32_16x16x32_f16 with 16-pixel
// tiles as the N dimension.  f16 operands / f32 accumulation.  B tiles are
// stored transposed [n][k] (stride 40 halves: 16B-aligned, bank-conflict-free)
// so fragment builds are ds_load_b128.  Where available, the f_k key tiles are
// fetched with the Tensor Data Mover (tensor_load_to_lds + s_wait_tensorcnt).
// ---------------------------------------------------------------------------

#define NIMG 16
#define HWPX (240 * 240)        // 57600
#define HW2  (120 * 120)        // 14400
#define BST  40                 // B-tile row stride in halves
#define FST  20                 // f32 staging row stride in floats

typedef __attribute__((ext_vector_type(16))) _Float16 v16h;
typedef __attribute__((ext_vector_type(8)))  _Float16 v8h;
typedef __attribute__((ext_vector_type(8)))  float    v8f;
typedef __attribute__((ext_vector_type(4)))  float    v4f;

#define WMMA_F16(a, b, c) \
  __builtin_amdgcn_wmma_f32_16x16x32_f16(false, (a), false, (b), (short)0, (c), false, false)

// ---------------- optional Tensor Data Mover path --------------------------
#if defined(__has_builtin)
#if __has_builtin(__builtin_amdgcn_tensor_load_to_lds)
#define HAVE_TDM 1
#endif
#endif
#ifndef HAVE_TDM
#define HAVE_TDM 0
#endif

#if HAVE_TDM
typedef unsigned int u32x4 __attribute__((ext_vector_type(4)));
typedef int          i32x8 __attribute__((ext_vector_type(8)));
typedef int          i32x4 __attribute__((ext_vector_type(4)));

// DMA one 16x16 f16 tile (row stride HWPX elements) from global into LDS,
// packed row-major [k][px] (32B rows).  D# words per CDNA5 ISA §8.3/8.4.
__device__ __forceinline__ void tdm_load_tile16(const _Float16* gsrc, unsigned lds_off) {
  const unsigned long long ga = (unsigned long long)(size_t)(const void*)gsrc;
  u32x4 g0 = { 1u,                                   // count=1, is_restore=0
               lds_off,                               // lds_addr (bytes)
               (unsigned)ga,                          // global_addr[31:0]
               ((unsigned)(ga >> 32) & 0x01FFFFFFu) | (2u << 30) };  // addr[56:32] | type=2
  i32x8 g1;
  g1[0] = 1 << 16;                                    // data_size = 2B
  g1[1] = (int)(((unsigned)HWPX & 0xFFFFu) << 16);    // tensor_dim0[15:0]
  g1[2] = (int)((((unsigned)HWPX >> 16) & 0xFFFFu) |  // tensor_dim0[31:16]
                (16u << 16));                         // tensor_dim1[15:0] = 16
  g1[3] = (int)(16u << 16);                           // tile_dim0 = 16
  g1[4] = 16;                                         // tile_dim1 = 16, tile_dim2 = 0
  g1[5] = (int)(unsigned)HWPX;                        // tensor_dim0_stride[31:0]
  g1[6] = 0;                                          // stride hi / dim1_stride (unused, 2D)
  g1[7] = 0;
  i32x4 gz = { 0, 0, 0, 0 };
#if __has_include(<hip/amd_detail/amd_gfx1250_TDM.h>)
  i32x8 gz8 = { 0, 0, 0, 0, 0, 0, 0, 0 };
  __builtin_amdgcn_tensor_load_to_lds(g0, g1, gz, gz, gz8, 0);
#else
  __builtin_amdgcn_tensor_load_to_lds(g0, g1, gz, gz, 0);
#endif
}

__device__ __forceinline__ void wait_tensorcnt0() {
#if __has_builtin(__builtin_amdgcn_s_wait_tensorcnt)
  __builtin_amdgcn_s_wait_tensorcnt(0);
#else
  asm volatile("s_wait_tensorcnt 0x0" ::: "memory");
#endif
}

// B fragment (32x16) from TDM row-major tile [k][px] (16 real rows, k>=16 == 0).
__device__ __forceinline__ v16h frag_b_rm16(const _Float16* X, int lane) {
  const int g = lane >> 4, n = lane & 15;
  v16h b;
#pragma unroll
  for (int j = 0; j < 16; ++j) b[j] = (g == 0) ? X[j * 16 + n] : (_Float16)0.f;
  return b;
}
#endif  // HAVE_TDM

// A fragment (16x32 f16), source row-major W[m*ldk + k], 16B-aligned rows.
__device__ __forceinline__ v16h frag_a(const _Float16* W, int ldk, int lane) {
  const int g = lane >> 4, m = lane & 15;
  const v8h lo = *(const v8h*)(W + m * ldk + 16 * g);
  const v8h hi = *(const v8h*)(W + m * ldk + 16 * g + 8);
  return __builtin_shufflevector(lo, hi, 0, 1, 2, 3, 4, 5, 6, 7,
                                 8, 9, 10, 11, 12, 13, 14, 15);
}

// B fragment (32x16 f16) from transposed storage Xt[n*BST + k].
__device__ __forceinline__ v16h frag_b(const _Float16* Xt, int lane) {
  const int g = lane >> 4, n = lane & 15;
  const v8h lo = *(const v8h*)(Xt + n * BST + 16 * g);
  const v8h hi = *(const v8h*)(Xt + n * BST + 16 * g + 8);
  return __builtin_shufflevector(lo, hi, 0, 1, 2, 3, 4, 5, 6, 7,
                                 8, 9, 10, 11, 12, 13, 14, 15);
}
// D/C layout: VGPR v holds row m = v + 8*(lane>>4), col n = lane&15.

// ---------------------------------------------------------------------------
__global__ void kZero(float* p, int n) {
  int i = blockIdx.x * 256 + threadIdx.x;
  if (i < n) p[i] = 0.f;
}

// ---------------------------------------------------------------------------
// Kernel A: Tc -> softmax att (channel 1), f1_k / f2_k (stored f16),
//           adaptive-pool partial sums of f_k * att.
// ---------------------------------------------------------------------------
__global__ __launch_bounds__(256) void kA(
    const float* __restrict__ T1, const float* __restrict__ T2,
    const float* __restrict__ w_fc, const float* __restrict__ b_fc,
    const float* __restrict__ w_k,  const float* __restrict__ b_k,
    _Float16* __restrict__ f1k, _Float16* __restrict__ f2k,
    float* __restrict__ vsum1, float* __restrict__ vsum2) {
  __shared__ alignas(16) _Float16 sWfc[16 * 64];
  __shared__ alignas(16) _Float16 sWk[16 * 32];
  __shared__ float sBfc[16], sBk[16];
  __shared__ alignas(16) _Float16 sXa[8][16 * BST];  // T1 tile  [col][k=c]
  __shared__ alignas(16) _Float16 sXb[8][16 * BST];  // T2u tile [col][k=c]
  __shared__ alignas(16) float sTc[8][16 * FST];     // [col][m]
  __shared__ float sAtt[8][16];
  __shared__ float sVacc[192];

  const int tid = threadIdx.x;
  const int wave = tid >> 5, lane = tid & 31;
  const int g = lane >> 4, col = lane & 15;

  for (int i = tid; i < 16 * 64; i += 256) sWfc[i] = (_Float16)w_fc[i];
  for (int i = tid; i < 16 * 32; i += 256) sWk[i] = (_Float16)w_k[i];
  if (tid < 16) { sBfc[tid] = b_fc[tid]; sBk[tid] = b_k[tid]; }
  for (int i = tid; i < 192; i += 256) sVacc[i] = 0.f;
  __syncthreads();

  const long pix0 = (long)blockIdx.x * 128 + wave * 16;
  const int n = (int)(pix0 / HWPX);
  const int rem = (int)(pix0 % HWPX);
  const int h = rem / 240, w0 = rem % 240;

  // stage tiles: lane (g,col) owns channels 16g..16g+15 at pixel col
  {
    const float* s1 = T1 + ((size_t)(n * 32 + 16 * g)) * HWPX + rem + col;
    const float* s2 = T2 + ((size_t)(n * 32 + 16 * g)) * HW2 +
                      (size_t)(h >> 1) * 120 + ((w0 + col) >> 1);
    v8h a0, a1, b0, b1;
#pragma unroll
    for (int i = 0; i < 8; ++i) {
      a0[i] = (_Float16)s1[(size_t)i * HWPX];
      a1[i] = (_Float16)s1[(size_t)(i + 8) * HWPX];
      b0[i] = (_Float16)s2[(size_t)i * HW2];
      b1[i] = (_Float16)s2[(size_t)(i + 8) * HW2];
    }
    *(v8h*)(sXa[wave] + col * BST + 16 * g)     = a0;
    *(v8h*)(sXa[wave] + col * BST + 16 * g + 8) = a1;
    *(v8h*)(sXb[wave] + col * BST + 16 * g)     = b0;
    *(v8h*)(sXb[wave] + col * BST + 16 * g + 8) = b1;
  }

  const v16h aFc0 = frag_a(sWfc, 64, lane);       // k 0..31  (T1 part)
  const v16h aFc1 = frag_a(sWfc + 32, 64, lane);  // k 32..63 (T2u part)
  const v16h aK   = frag_a(sWk, 32, lane);
  const v16h bA = frag_b(sXa[wave], lane);
  const v16h bB = frag_b(sXb[wave], lane);

  v8f tc = {}; tc = WMMA_F16(aFc0, bA, tc); tc = WMMA_F16(aFc1, bB, tc);
  v8f k1 = {}; k1 = WMMA_F16(aK, bA, k1);
  v8f k2 = {}; k2 = WMMA_F16(aK, bB, k2);

  float fk1v[8], fk2v[8];
  v4f t0, t1;
#pragma unroll
  for (int v = 0; v < 8; ++v) {
    const int m = v + 8 * g;
    fk1v[v] = k1[v] + sBk[m];
    fk2v[v] = k2[v] + sBk[m];
    f1k[((size_t)n * 16 + m) * HWPX + rem + col] = (_Float16)fk1v[v];
    f2k[((size_t)n * 16 + m) * HWPX + rem + col] = (_Float16)fk2v[v];
    if (v < 4) t0[v] = tc[v] + sBfc[m]; else t1[v - 4] = tc[v] + sBfc[m];
  }
  *(v4f*)(sTc[wave] + col * FST + 8 * g)     = t0;
  *(v4f*)(sTc[wave] + col * FST + 8 * g + 4) = t1;

  if (lane < 16) {             // per-pixel softmax over 16 channels, take ch 1
    float mx = -1e30f;
#pragma unroll
    for (int m = 0; m < 16; ++m) mx = fmaxf(mx, sTc[wave][lane * FST + m]);
    float sum = 0.f, e1 = 0.f;
#pragma unroll
    for (int m = 0; m < 16; ++m) {
      float e = __expf(sTc[wave][lane * FST + m] - mx);
      sum += e;
      if (m == 1) e1 = e;
    }
    sAtt[wave][lane] = e1 / sum;
  }
  const float att = sAtt[wave][col];

  const int bx = (w0 + col) / 40;
#pragma unroll
  for (int v = 0; v < 8; ++v) {
    const int m = v + 8 * g;
    atomicAdd(&sVacc[(0 * 16 + m) * 6 + bx], fk1v[v] * att);
    atomicAdd(&sVacc[(1 * 16 + m) * 6 + bx], fk2v[v] * att);
  }
  __syncthreads();

  const long p0b = (long)blockIdx.x * 128;   // 40-row pool bands align with blocks
  const int nb = (int)(p0b / HWPX);
  const int by0 = ((int)(p0b % HWPX) / 240) / 40;
  for (int i = tid; i < 192; i += 256) {
    const int brm = i / 6, bxi = i % 6;
    const int br = brm >> 4, m = brm & 15;
    float* dst = br ? vsum2 : vsum1;
    atomicAdd(&dst[((size_t)nb * 16 + m) * 36 + by0 * 6 + bxi], sVacc[i]);
  }
}

// ---------------------------------------------------------------------------
// Kernel B: finalize V = inner 4x4 of 6x6 pool / 1600.
// ---------------------------------------------------------------------------
__global__ void kB(const float* __restrict__ vsum1, const float* __restrict__ vsum2,
                   float* __restrict__ V1, float* __restrict__ V2) {
  int idx = blockIdx.x * 256 + threadIdx.x;
  if (idx >= 2 * 16 * 16 * 16) return;
  const int j = idx & 15, s = (idx >> 4) & 15, n = (idx >> 8) & 15, br = idx >> 12;
  const int by = 1 + (j >> 2), bx = 1 + (j & 3);
  const float* src = br ? vsum2 : vsum1;
  float* dst = br ? V2 : V1;
  dst[((size_t)n * 16 + s) * 16 + j] = src[((size_t)n * 16 + s) * 36 + by * 6 + bx] * (1.f / 1600.f);
}

// ---------------------------------------------------------------------------
// Kernel C: proj = softmax(V^T K), Q = Wq x, n_state += Q proj^T.
// ---------------------------------------------------------------------------
__global__ __launch_bounds__(256) void kC(
    const float* __restrict__ T1, const float* __restrict__ T2,
    const _Float16* __restrict__ f1k, const _Float16* __restrict__ f2k,
    const float* __restrict__ V1, const float* __restrict__ V2,
    const float* __restrict__ w_f1q, const float* __restrict__ b_f1q,
    const float* __restrict__ w_f2q, const float* __restrict__ b_f2q,
    float* __restrict__ ns1, float* __restrict__ ns2) {
  __shared__ alignas(16) _Float16 sVt[2][512];
  __shared__ alignas(16) _Float16 sWq[2][512];
  __shared__ float sBq[2][16];
#if HAVE_TDM
  __shared__ alignas(16) _Float16 sKr[8][2][256];    // TDM tiles [k][px]
#else
  __shared__ alignas(16) _Float16 sK[8][16 * BST];   // [col][k], k>=16 zero
#endif
  __shared__ alignas(16) _Float16 sXc[8][16 * BST];  // [col][k=c]
  __shared__ alignas(16) float sL[8][16 * FST];      // [col][m]
  __shared__ alignas(16) _Float16 sP[8][16 * BST];   // [m][k=px], k>=16 zero
  __shared__ alignas(16) _Float16 sQa[8][512];       // A: [s][k=px], k>=16 zero
  __shared__ float nsAcc[512];

  const int tid = threadIdx.x;
  const int wave = tid >> 5, lane = tid & 31;
  const int g = lane >> 4, col = lane & 15;

  const long p0b = (long)blockIdx.x * 128;
  const int n = (int)(p0b / HWPX);

  for (int i = tid; i < 512; i += 256) {
    const int m = i >> 5, k = i & 31;
    sVt[0][i] = (k < 16) ? (_Float16)V1[((size_t)n * 16 + k) * 16 + m] : (_Float16)0.f;
    sVt[1][i] = (k < 16) ? (_Float16)V2[((size_t)n * 16 + k) * 16 + m] : (_Float16)0.f;
    sWq[0][i] = (_Float16)w_f1q[i];
    sWq[1][i] = (_Float16)w_f2q[i];
    nsAcc[i] = 0.f;
  }
  if (tid < 16) { sBq[0][tid] = b_f1q[tid]; sBq[1][tid] = b_f2q[tid]; }
  __syncthreads();

  const long pix0 = p0b + wave * 16;
  const int rem = (int)(pix0 % HWPX);
  const int h = rem / 240, w0 = rem % 240;

  // zero pads once
  {
    const v8h z = {};
#if !HAVE_TDM
    *(v8h*)(sK[wave] + col * BST + 16 + 8 * g) = z;       // k 16..31
#endif
    if (lane < 16) {                                      // sP rows k 16..31
      *(v8h*)(sP[wave] + lane * BST + 16) = z;
      *(v8h*)(sP[wave] + lane * BST + 24) = z;
    }
    for (int i = lane; i < 512; i += 32)
      if ((i & 31) >= 16) sQa[wave][i] = (_Float16)0.f;
  }

#if HAVE_TDM
  // async DMA both branches' key tiles, then one wait
  tdm_load_tile16(f1k + ((size_t)n * 16) * HWPX + rem,
                  (unsigned)(size_t)(const void*)&sKr[wave][0][0]);
  tdm_load_tile16(f2k + ((size_t)n * 16) * HWPX + rem,
                  (unsigned)(size_t)(const void*)&sKr[wave][1][0]);
  wait_tensorcnt0();
#endif

  __builtin_prefetch(T1 + ((size_t)(n * 32 + 16 * g)) * HWPX + rem + col, 0, 1);

  for (int br = 0; br < 2; ++br) {
#if HAVE_TDM
    const v16h bK = frag_b_rm16(&sKr[wave][br][0], lane);
#else
    {
      const _Float16* fkc = (br ? f2k : f1k) + ((size_t)n * 16 + 8 * g) * HWPX + rem + col;
      v8h kv;
#pragma unroll
      for (int i = 0; i < 8; ++i) kv[i] = fkc[(size_t)i * HWPX];
      *(v8h*)(sK[wave] + col * BST + 8 * g) = kv;
    }
    const v16h bK = frag_b(sK[wave], lane);
#endif

    v8f lg = {};
    lg = WMMA_F16(frag_a(sVt[br], 32, lane), bK, lg);
    {
      v4f l0, l1;
#pragma unroll
      for (int v = 0; v < 8; ++v) { if (v < 4) l0[v] = lg[v]; else l1[v - 4] = lg[v]; }
      *(v4f*)(sL[wave] + col * FST + 8 * g)     = l0;
      *(v4f*)(sL[wave] + col * FST + 8 * g + 4) = l1;
    }

    if (lane < 16) {                             // softmax over node dim m
      float mx = -1e30f;
#pragma unroll
      for (int m = 0; m < 16; ++m) mx = fmaxf(mx, sL[wave][lane * FST + m]);
      float pe[16], sum = 0.f;
#pragma unroll
      for (int m = 0; m < 16; ++m) { pe[m] = __expf(sL[wave][lane * FST + m] - mx); sum += pe[m]; }
      const float inv = 1.f / sum;
#pragma unroll
      for (int m = 0; m < 16; ++m) sP[wave][m * BST + lane] = (_Float16)(pe[m] * inv);
    }

    // X tile for Q: lane owns channels 16g..16g+15
    {
      v8h x0, x1;
      if (br == 0) {
        const float* s1 = T1 + ((size_t)(n * 32 + 16 * g)) * HWPX + rem + col;
#pragma unroll
        for (int i = 0; i < 8; ++i) {
          x0[i] = (_Float16)s1[(size_t)i * HWPX];
          x1[i] = (_Float16)s1[(size_t)(i + 8) * HWPX];
        }
      } else {
        const float* s2 = T2 + ((size_t)(n * 32 + 16 * g)) * HW2 +
                          (size_t)(h >> 1) * 120 + ((w0 + col) >> 1);
#pragma unroll
        for (int i = 0; i < 8; ++i) {
          x0[i] = (_Float16)s2[(size_t)i * HW2];
          x1[i] = (_Float16)s2[(size_t)(i + 8) * HW2];
        }
      }
      *(v8h*)(sXc[wave] + col * BST + 16 * g)     = x0;
      *(v8h*)(sXc[wave] + col * BST + 16 * g + 8) = x1;
    }
    v8f q = {};
    q = WMMA_F16(frag_a(sWq[br], 32, lane), frag_b(sXc[wave], lane), q);
#pragma unroll
    for (int v = 0; v < 8; ++v)
      sQa[wave][(v + 8 * g) * 32 + col] = (_Float16)(q[v] + sBq[br][v + 8 * g]);  // A[s][k=px]

    v8f nsd = {};
    nsd = WMMA_F16(frag_a(sQa[wave], 32, lane), frag_b(sP[wave], lane), nsd);
#pragma unroll
    for (int v = 0; v < 8; ++v)
      atomicAdd(&nsAcc[br * 256 + (v + 8 * g) * 16 + col], nsd[v]);
  }
  __syncthreads();
  for (int i = tid; i < 512; i += 256) {
    float* dst = (i < 256) ? ns1 : ns2;
    atomicAdd(&dst[(size_t)n * 256 + (i & 255)], nsAcc[i]);
  }
}

// ---------------------------------------------------------------------------
// Kernel D: GCN on 16x16 node states (tiny; scalar VALU).
// ---------------------------------------------------------------------------
__global__ __launch_bounds__(256) void kD(
    const float* __restrict__ ns1, const float* __restrict__ ns2,
    const float* __restrict__ g1w1, const float* __restrict__ g1b1, const float* __restrict__ g1w2,
    const float* __restrict__ g2w1, const float* __restrict__ g2b1, const float* __restrict__ g2w2,
    float* __restrict__ nr1, float* __restrict__ nr2) {
  __shared__ float hb[256];
  const int tid = threadIdx.x;
  const int s = tid >> 4, m = tid & 15;
  for (int br = 0; br < 2; ++br) {
    const float* ns = br ? ns2 : ns1;
    const float* w1 = br ? g2w1 : g1w1;
    const float* b1 = br ? g2b1 : g1b1;
    const float* w2 = br ? g2w2 : g1w2;
    float* nr = br ? nr2 : nr1;
    for (int n = 0; n < 16; ++n) {
      const float* x = ns + n * 256;
      float acc = b1[m];
#pragma unroll
      for (int i = 0; i < 16; ++i) acc += w1[m * 16 + i] * x[s * 16 + i];
      acc -= x[s * 16 + m];
      __syncthreads();
      hb[tid] = acc;
      __syncthreads();
      float o = 0.f;
#pragma unroll
      for (int i = 0; i < 16; ++i) o += w2[s * 16 + i] * hb[i * 16 + m];
      nr[n * 256 + tid] = fmaxf(o, 0.f);
    }
  }
}

// ---------------------------------------------------------------------------
// Kernel E: proj recompute, state = n_rel*proj, residual + ext conv,
//           final 32x64 output conv.  Writes d_out.
// ---------------------------------------------------------------------------
__global__ __launch_bounds__(256) void kE(
    const float* __restrict__ T1, const float* __restrict__ T2,
    const _Float16* __restrict__ f1k, const _Float16* __restrict__ f2k,
    const float* __restrict__ V1, const float* __restrict__ V2,
    const float* __restrict__ nr1, const float* __restrict__ nr2,
    const float* __restrict__ w_f1ext, const float* __restrict__ w_f2ext,
    const float* __restrict__ w_out, const float* __restrict__ b_out,
    float* __restrict__ out) {
  __shared__ alignas(16) _Float16 sVt[2][512];
  __shared__ alignas(16) _Float16 sNr[2][512];
  __shared__ alignas(16) _Float16 sE[2][1024];
  __shared__ alignas(16) _Float16 sWo[2048];
  __shared__ float sBo[32];
#if HAVE_TDM
  __shared__ alignas(16) _Float16 sKr[8][2][256];    // TDM tiles [k][px]
#else
  __shared__ alignas(16) _Float16 sK[8][16 * BST];   // [col][k], k>=16 zero
#endif
  __shared__ alignas(16) float sL[8][16 * FST];      // [col][m]
  __shared__ alignas(16) _Float16 sPS[8][16 * BST];  // proj [px][m] then state [px][s]
  __shared__ alignas(16) _Float16 sF1[8][16 * BST];  // [px][k=c], full 32
  __shared__ alignas(16) _Float16 sF2[8][16 * BST];

  const int tid = threadIdx.x;
  const int wave = tid >> 5, lane = tid & 31;
  const int g = lane >> 4, col = lane & 15;

  const long p0b = (long)blockIdx.x * 128;
  const int n = (int)(p0b / HWPX);

  for (int i = tid; i < 2048; i += 256) {
    sWo[i] = (_Float16)w_out[i];
    if (i < 512) {
      const int m = i >> 5, k = i & 31;
      sVt[0][i] = (k < 16) ? (_Float16)V1[((size_t)n * 16 + k) * 16 + m] : (_Float16)0.f;
      sVt[1][i] = (k < 16) ? (_Float16)V2[((size_t)n * 16 + k) * 16 + m] : (_Float16)0.f;
      sNr[0][i] = (k < 16) ? (_Float16)nr1[(size_t)n * 256 + m * 16 + k] : (_Float16)0.f;
      sNr[1][i] = (k < 16) ? (_Float16)nr2[(size_t)n * 256 + m * 16 + k] : (_Float16)0.f;
    }
    if (i < 1024) {
      const int r = i >> 5, k = i & 31;
      sE[0][i] = (k < 16) ? (_Float16)w_f1ext[r * 16 + k] : (_Float16)0.f;
      sE[1][i] = (k < 16) ? (_Float16)w_f2ext[r * 16 + k] : (_Float16)0.f;
    }
  }
  if (tid < 32) sBo[tid] = b_out[tid];
  __syncthreads();

  const long pix0 = p0b + wave * 16;
  const int rem = (int)(pix0 % HWPX);
  const int h = rem / 240, w0 = rem % 240;

  {
    const v8h z = {};
#if !HAVE_TDM
    *(v8h*)(sK[wave] + col * BST + 16 + 8 * g) = z;    // k 16..31
#endif
    *(v8h*)(sPS[wave] + col * BST + 16 + 8 * g) = z;   // k 16..31
  }

#if HAVE_TDM
  tdm_load_tile16(f1k + ((size_t)n * 16) * HWPX + rem,
                  (unsigned)(size_t)(const void*)&sKr[wave][0][0]);
  tdm_load_tile16(f2k + ((size_t)n * 16) * HWPX + rem,
                  (unsigned)(size_t)(const void*)&sKr[wave][1][0]);
  wait_tensorcnt0();
#endif

  __builtin_prefetch(T1 + ((size_t)(n * 32 + 8 * g)) * HWPX + rem + col, 0, 1);
  __builtin_prefetch(T2 + ((size_t)(n * 32 + 8 * g)) * HW2 +
                     (size_t)(h >> 1) * 120 + ((w0 + col) >> 1), 0, 1);

  for (int br = 0; br < 2; ++br) {
#if HAVE_TDM
    const v16h bK = frag_b_rm16(&sKr[wave][br][0], lane);
#else
    {
      const _Float16* fkc = (br ? f2k : f1k) + ((size_t)n * 16 + 8 * g) * HWPX + rem + col;
      v8h kv;
#pragma unroll
      for (int i = 0; i < 8; ++i) kv[i] = fkc[(size_t)i * HWPX];
      *(v8h*)(sK[wave] + col * BST + 8 * g) = kv;
    }
    const v16h bK = frag_b(sK[wave], lane);
#endif

    v8f lg = {};
    lg = WMMA_F16(frag_a(sVt[br], 32, lane), bK, lg);
    {
      v4f l0, l1;
#pragma unroll
      for (int v = 0; v < 8; ++v) { if (v < 4) l0[v] = lg[v]; else l1[v - 4] = lg[v]; }
      *(v4f*)(sL[wave] + col * FST + 8 * g)     = l0;
      *(v4f*)(sL[wave] + col * FST + 8 * g + 4) = l1;
    }

    if (lane < 16) {                 // softmax over m; proj -> sPS [px][m]
      float mx = -1e30f;
#pragma unroll
      for (int m = 0; m < 16; ++m) mx = fmaxf(mx, sL[wave][lane * FST + m]);
      float pe[16], sum = 0.f;
#pragma unroll
      for (int m = 0; m < 16; ++m) { pe[m] = __expf(sL[wave][lane * FST + m] - mx); sum += pe[m]; }
      const float inv = 1.f / sum;
      v8h p0, p1;
#pragma unroll
      for (int m = 0; m < 8; ++m) {
        p0[m] = (_Float16)(pe[m] * inv);
        p1[m] = (_Float16)(pe[m + 8] * inv);
      }
      *(v8h*)(sPS[wave] + lane * BST)     = p0;
      *(v8h*)(sPS[wave] + lane * BST + 8) = p1;
    }

    v8f st = {};
    st = WMMA_F16(frag_a(sNr[br], 32, lane), frag_b(sPS[wave], lane), st);
    {
      v8h sv;                         // state -> sPS [px][s] (overwrites proj)
#pragma unroll
      for (int v = 0; v < 8; ++v) sv[v] = (_Float16)st[v];
      *(v8h*)(sPS[wave] + col * BST + 8 * g) = sv;
    }

    v8f e0 = {}, e1 = {};
    e0 = WMMA_F16(frag_a(sE[br], 32, lane), frag_b(sPS[wave], lane), e0);
    e1 = WMMA_F16(frag_a(sE[br] + 512, 32, lane), frag_b(sPS[wave], lane), e1);

    _Float16* sF = br ? sF2[wave] : sF1[wave];
    {
      v8h f0v, f1v;
#pragma unroll
      for (int v = 0; v < 8; ++v) {
        const int c0 = v + 8 * g, c1 = c0 + 16;
        float r0, r1;
        if (br == 0) {
          r0 = T1[(size_t)(n * 32 + c0) * HWPX + rem + col];
          r1 = T1[(size_t)(n * 32 + c1) * HWPX + rem + col];
        } else {
          const size_t b2 = (size_t)(h >> 1) * 120 + ((w0 + col) >> 1);
          r0 = T2[(size_t)(n * 32 + c0) * HW2 + b2];
          r1 = T2[(size_t)(n * 32 + c1) * HW2 + b2];
        }
        f0v[v] = (_Float16)(r0 + e0[v]);
        f1v[v] = (_Float16)(r1 + e1[v]);
      }
      *(v8h*)(sF + col * BST + 8 * g)      = f0v;   // k = c0 range
      *(v8h*)(sF + col * BST + 16 + 8 * g) = f1v;   // k = c1 range
    }
  }

  v8f o0 = {}, o1 = {};
  o0 = WMMA_F16(frag_a(sWo, 64, lane),                 frag_b(sF1[wave], lane), o0);
  o0 = WMMA_F16(frag_a(sWo + 32, 64, lane),            frag_b(sF2[wave], lane), o0);
  o1 = WMMA_F16(frag_a(sWo + 16 * 64, 64, lane),       frag_b(sF1[wave], lane), o1);
  o1 = WMMA_F16(frag_a(sWo + 16 * 64 + 32, 64, lane),  frag_b(sF2[wave], lane), o1);
#pragma unroll
  for (int v = 0; v < 8; ++v) {
    const int c0 = v + 8 * g;
    out[(size_t)(n * 32 + c0) * HWPX + rem + col]      = o0[v] + sBo[c0];
    out[(size_t)(n * 32 + c0 + 16) * HWPX + rem + col] = o1[v] + sBo[c0 + 16];
  }
}

// ---------------------------------------------------------------------------
extern "C" void kernel_launch(void* const* d_in, const int* in_sizes, int n_in,
                              void* d_out, int out_size, void* d_ws, size_t ws_size,
                              hipStream_t stream) {
  (void)in_sizes; (void)n_in; (void)out_size; (void)ws_size;
  const float* T1      = (const float*)d_in[0];
  const float* T2      = (const float*)d_in[1];
  const float* w_fc    = (const float*)d_in[2];
  const float* b_fc    = (const float*)d_in[3];
  const float* w_out   = (const float*)d_in[4];
  const float* b_out   = (const float*)d_in[5];
  const float* w_f1q   = (const float*)d_in[6];
  const float* b_f1q   = (const float*)d_in[7];
  const float* w_f1k   = (const float*)d_in[8];
  const float* b_f1k   = (const float*)d_in[9];
  const float* w_f2q   = (const float*)d_in[10];
  const float* b_f2q   = (const float*)d_in[11];
  const float* g1_w1   = (const float*)d_in[12];
  const float* g1_b1   = (const float*)d_in[13];
  const float* g1_w2   = (const float*)d_in[14];
  const float* w_f1ext = (const float*)d_in[15];
  const float* g2_w1   = (const float*)d_in[16];
  const float* g2_b1   = (const float*)d_in[17];
  const float* g2_w2   = (const float*)d_in[18];
  const float* w_f2ext = (const float*)d_in[19];

  // Workspace layout: f16 key tensors, then small f32 accumulators.
  const size_t fkE = (size_t)16 * 16 * HWPX;
  _Float16* f1k = (_Float16*)d_ws;
  _Float16* f2k = f1k + fkE;
  float* fb     = (float*)(f2k + fkE);
  float* vsum1  = fb;                 // 16*16*36
  float* vsum2  = vsum1 + 9216;
  float* V1     = vsum2 + 9216;       // 16*16*16
  float* V2     = V1 + 4096;
  float* ns1    = V2 + 4096;
  float* ns2    = ns1 + 4096;
  float* nr1    = ns2 + 4096;
  float* nr2    = nr1 + 4096;

  const int zeroN = 9216 * 2 + 4096 * 6;   // zero accumulators every launch
  kZero<<<(zeroN + 255) / 256, 256, 0, stream>>>(vsum1, zeroN);

  const int nBlocks = (NIMG * HWPX) / 128;  // 7200
  kA<<<nBlocks, 256, 0, stream>>>(T1, T2, w_fc, b_fc, w_f1k, b_f1k, f1k, f2k, vsum1, vsum2);
  kB<<<32, 256, 0, stream>>>(vsum1, vsum2, V1, V2);
  kC<<<nBlocks, 256, 0, stream>>>(T1, T2, f1k, f2k, V1, V2, w_f1q, b_f1q, w_f2q, b_f2q, ns1, ns2);
  kD<<<1, 256, 0, stream>>>(ns1, ns2, g1_w1, g1_b1, g1_w2, g2_w1, g2_b1, g2_w2, nr1, nr2);
  kE<<<nBlocks, 256, 0, stream>>>(T1, T2, f1k, f2k, V1, V2, nr1, nr2,
                                  w_f1ext, w_f2ext, w_out, b_out, (float*)d_out);
}
